// EISANImodel_13941463843069
// MI455X (gfx1250) — 8-bit matrix operand order repaired and checked by reading, compiled, not run on hardware
//
#include <hip/hip_runtime.h>
#include <stdint.h>


typedef __attribute__((ext_vector_type(8))) int v8i;

#define B_ROWS 512
#define F_IN   128
#define NBITS  16
#define E_ENC  2048
#define HID    8192
#define NCLS   10
#define THRESH_I 4

__global__ void zero_f32(float* __restrict__ p, int n) {
  int i = blockIdx.x * blockDim.x + threadIdx.x;
  if (i < n) p[i] = 0.0f;
}

__global__ void convert_w_i8(const float* __restrict__ w,
                             int8_t* __restrict__ o, int n4) {
  int stride = gridDim.x * blockDim.x;
  for (int i = blockIdx.x * blockDim.x + threadIdx.x; i < n4; i += stride) {
    float4 v = ((const float4*)w)[i];
    uint32_t b0 = (uint8_t)(int8_t)((v.x > 0.5f) ? 1 : ((v.x < -0.5f) ? -1 : 0));
    uint32_t b1 = (uint8_t)(int8_t)((v.y > 0.5f) ? 1 : ((v.y < -0.5f) ? -1 : 0));
    uint32_t b2 = (uint8_t)(int8_t)((v.z > 0.5f) ? 1 : ((v.z < -0.5f) ? -1 : 0));
    uint32_t b3 = (uint8_t)(int8_t)((v.w > 0.5f) ? 1 : ((v.w < -0.5f) ? -1 : 0));
    ((uint32_t*)o)[i] = b0 | (b1 << 8) | (b2 << 16) | (b3 << 24);
  }
}

__global__ void encode_thermo(const float* __restrict__ x,
                              uint8_t* __restrict__ act0) {
  int i = blockIdx.x * blockDim.x + threadIdx.x;
  if (i >= B_ROWS * F_IN) return;
  float v = x[i] * (float)NBITS;
  uint32_t w[4] = {0u, 0u, 0u, 0u};
#pragma unroll
  for (int j = 0; j < NBITS; ++j)
    if (v > (float)j) w[j >> 2] |= 1u << ((j & 3) * 8);
  ((uint4*)act0)[i] = make_uint4(w[0], w[1], w[2], w[3]);
}

__device__ __forceinline__ v8i load_a(const uint8_t* __restrict__ p) {
  int2 a0 = *(const int2*)(p + 0);
  int2 a1 = *(const int2*)(p + 16);
  int2 a2 = *(const int2*)(p + 32);
  int2 a3 = *(const int2*)(p + 48);
  return (v8i){a0.x, a0.y, a1.x, a1.y, a2.x, a2.y, a3.x, a3.y};
}
__device__ __forceinline__ v8i load_b(const int8_t* __restrict__ p) {
  int2 q0 = *(const int2*)(p + 0);
  int2 q1 = *(const int2*)(p + 16);
  int2 q2 = *(const int2*)(p + 32);
  int2 q3 = *(const int2*)(p + 48);
  return (v8i){q0.x, q0.y, q1.x, q1.y, q2.x, q2.y, q3.x, q3.y};
}

__device__ __forceinline__ v8i wmma_iu8(v8i a, v8i b, v8i c) {
  return __builtin_amdgcn_wmma_i32_16x16x64_iu8(false, a, true, b, c,
                                                false, false);
}

__global__ __launch_bounds__(128) void gemm_iu8_thresh(
    const uint8_t* __restrict__ A, const int8_t* __restrict__ W,
    uint8_t* __restrict__ out, int K, int H) {
  const int lane = threadIdx.x & 31;
  const int wv   = threadIdx.x >> 5;
  const int ln   = lane & 15;
  const int lh   = lane >> 4;
  const int mTile = blockIdx.x * 128 + wv * 32;
  const int nTile = blockIdx.y * 64;

  v8i acc[2][4];
#pragma unroll
  for (int mt = 0; mt < 2; ++mt)
#pragma unroll
    for (int t = 0; t < 4; ++t) acc[mt][t] = (v8i){0, 0, 0, 0, 0, 0, 0, 0};

  const uint8_t* aPtr0 = A + (size_t)(mTile + 0  + ln) * K + (size_t)lh * 8;
  const uint8_t* aPtr1 = A + (size_t)(mTile + 16 + ln) * K + (size_t)lh * 8;
  const int8_t* bPtr[4];
#pragma unroll
  for (int t = 0; t < 4; ++t)
    bPtr[t] = W + (size_t)(nTile + t * 16 + ln) * K + (size_t)lh * 8;

  v8i a0 = load_a(aPtr0);
  v8i a1 = load_a(aPtr1);
  v8i bb[4];
#pragma unroll
  for (int t = 0; t < 4; ++t) bb[t] = load_b(bPtr[t]);

  for (int k = 64; k < K; k += 64) {
    v8i na0 = load_a(aPtr0 + k);
    v8i na1 = load_a(aPtr1 + k);
    v8i nb[4];
#pragma unroll
    for (int t = 0; t < 4; ++t) nb[t] = load_b(bPtr[t] + k);

#pragma unroll
    for (int t = 0; t < 4; ++t) {
      acc[0][t] = wmma_iu8(a0, bb[t], acc[0][t]);
      acc[1][t] = wmma_iu8(a1, bb[t], acc[1][t]);
    }
    a0 = na0;
    a1 = na1;
#pragma unroll
    for (int t = 0; t < 4; ++t) bb[t] = nb[t];
  }
#pragma unroll
  for (int t = 0; t < 4; ++t) {
    acc[0][t] = wmma_iu8(a0, bb[t], acc[0][t]);
    acc[1][t] = wmma_iu8(a1, bb[t], acc[1][t]);
  }

#pragma unroll
  for (int mt = 0; mt < 2; ++mt) {
#pragma unroll
    for (int t = 0; t < 4; ++t) {
      int n = nTile + t * 16 + ln;
#pragma unroll
      for (int i = 0; i < 8; ++i) {
        int m = mTile + mt * 16 + i + lh * 8;
        out[(size_t)m * H + n] =
            (acc[mt][t][i] >= THRESH_I) ? (uint8_t)1 : (uint8_t)0;
      }
    }
  }
}

__global__ __launch_bounds__(256) void out_accum(
    const uint8_t* __restrict__ act, const float* __restrict__ conn,
    float* __restrict__ outAct) {
  __shared__ float red[NCLS];
  const int b = blockIdx.x;
  const int tid = threadIdx.x;
  if (tid < NCLS) red[tid] = 0.0f;
  __syncthreads();
  float s[NCLS];
#pragma unroll
  for (int c = 0; c < NCLS; ++c) s[c] = 0.0f;
  const uint8_t* row = act + (size_t)b * HID;
  for (int h = tid; h < HID; h += 256) {
    if (row[h]) {
      const float* cp = conn + (size_t)h * NCLS;
#pragma unroll
      for (int c = 0; c < NCLS; ++c) s[c] += cp[c];
    }
  }
#pragma unroll
  for (int c = 0; c < NCLS; ++c) atomicAdd(&red[c], s[c]);
  __syncthreads();
  if (tid < NCLS) outAct[b * NCLS + tid] += red[tid];
}

__global__ void argmax_pred(float* __restrict__ d_out) {
  int b = blockIdx.x * blockDim.x + threadIdx.x;
  if (b >= B_ROWS) return;
  const float* row = d_out + B_ROWS + (size_t)b * NCLS;
  float best = row[0];
  int bi = 0;
#pragma unroll
  for (int c = 1; c < NCLS; ++c)
    if (row[c] > best) { best = row[c]; bi = c; }
  d_out[b] = (float)bi;
}

extern "C" void kernel_launch(void* const* d_in, const int* in_sizes, int n_in,
                              void* d_out, int out_size, void* d_ws,
                              size_t ws_size, hipStream_t stream) {
  const float* x       = (const float*)d_in[1];
  const float* W0      = (const float*)d_in[2];
  const float* W1      = (const float*)d_in[3];
  const float* W2      = (const float*)d_in[4];
  const float* outConn = (const float*)d_in[5];
  float* out = (float*)d_out;

  uint8_t* ws = (uint8_t*)d_ws;
  size_t off = 0;
  int8_t* w0i = (int8_t*)(ws + off); off += (size_t)HID * E_ENC;
  int8_t* w1i = (int8_t*)(ws + off); off += (size_t)HID * HID;
  int8_t* w2i = (int8_t*)(ws + off); off += (size_t)HID * HID;
  uint8_t* act0 = ws + off;          off += (size_t)B_ROWS * E_ENC;
  uint8_t* actA = ws + off;          off += (size_t)B_ROWS * HID;
  uint8_t* actB = ws + off;          off += (size_t)B_ROWS * HID;
  (void)ws_size; (void)in_sizes; (void)n_in; (void)out_size;

  const int nOut = B_ROWS + B_ROWS * NCLS;
  zero_f32<<<(nOut + 255) / 256, 256, 0, stream>>>(out, nOut);

  convert_w_i8<<<2048, 256, 0, stream>>>(W0, w0i, HID * E_ENC / 4);
  convert_w_i8<<<4096, 256, 0, stream>>>(W1, w1i, HID * HID / 4);
  convert_w_i8<<<4096, 256, 0, stream>>>(W2, w2i, HID * HID / 4);

  encode_thermo<<<(B_ROWS * F_IN + 255) / 256, 256, 0, stream>>>(x, act0);

  dim3 grid(B_ROWS / 128, HID / 64);
  float* outAct = out + B_ROWS;

  gemm_iu8_thresh<<<grid, 128, 0, stream>>>(act0, w0i, actA, E_ENC, HID);
  out_accum<<<B_ROWS, 256, 0, stream>>>(actA, outConn, outAct);

  gemm_iu8_thresh<<<grid, 128, 0, stream>>>(actA, w1i, actB, HID, HID);
  out_accum<<<B_ROWS, 256, 0, stream>>>(actB, outConn + (size_t)HID * NCLS, outAct);

  gemm_iu8_thresh<<<grid, 128, 0, stream>>>(actB, w2i, actA, HID, HID);
  out_accum<<<B_ROWS, 256, 0, stream>>>(actA, outConn + 2 * (size_t)HID * NCLS, outAct);

  argmax_pred<<<(B_ROWS + 255) / 256, 256, 0, stream>>>(out);
}
